// MomentumQueue_45054206935067
// MI455X (gfx1250) — matrix operand order repaired and checked by reading, not compiled, not run on hardware
//
#include <hip/hip_runtime.h>
#include <math.h>

// ---------------------------------------------------------------------------
// MomentumQueue retrieval for MI455X (gfx1250, wave32, WMMA).
//   query [4096,512] f32, queue [32768,512] f32 (unit rows)
//   out = retrieved[4096*512] ++ distances[4096] ++ weights[4096*16]
// GEMM: 137 GFLOP -> bf16 WMMA (v_wmma_f32_16x16x32_bf16), compute bound:
// inputs (72MB f32 + 32MB bf16) are L2-resident (192MB L2), HBM time ~3us.
// B-fragment stream is software-pipelined (half-tile double buffer) so the
// WMMA pipe never waits loadcnt==0; per-tile block barriers replaced with
// compiler-only fences (per-wave LDS staging + in-order DS pipe).
// Workspace: 4096*512*2 + 32768*512*2 = 37,748,736 bytes of d_ws.
// ---------------------------------------------------------------------------

typedef __attribute__((ext_vector_type(16))) __bf16 bf16x16;
typedef __attribute__((ext_vector_type(8)))  float  v8f;
typedef __attribute__((ext_vector_type(4)))  float  f32x4;

union ABFrag {
    f32x4   f[2];   // two 16B chunks (8 bf16 each)
    bf16x16 bf;     // WMMA operand view
};

#define B_ROWS   4096
#define Q_ROWS   32768
#define DIM      512
#define TOPK     16
#define NEG_INF  (-3.4e38f)

// Compiler-only memory fence: same-wave LDS ops are in-order in the DS pipe
// (DScnt), we only need to stop the compiler from reordering around it.
#define KFENCE() __asm__ volatile("" ::: "memory")

// float -> bf16 with round-to-nearest-even
__device__ __forceinline__ unsigned short f2bf(float f) {
    unsigned u = __float_as_uint(f);
    u += 0x7FFFu + ((u >> 16) & 1u);
    return (unsigned short)(u >> 16);
}

// Insert (v, gi) into a register-resident sorted-descending top-16 list.
__device__ __forceinline__ void insert16(float (&vals)[TOPK], int (&idxs)[TOPK],
                                         float v, int gi) {
    if (v <= vals[TOPK - 1]) return;   // threshold guard: common fast path
    float cv = v; int ci = gi;
#pragma unroll
    for (int p = 0; p < TOPK; ++p) {
        bool  gt = cv > vals[p];
        float tv = vals[p]; int ti = idxs[p];
        vals[p] = gt ? cv : tv;  idxs[p] = gt ? ci : ti;
        cv      = gt ? tv : cv;  ci      = gt ? ti : ci;
    }
}

// ---------------------------------------------------------------------------
// Kernel 1: L2-normalize each query row, store bf16. One wave per row.
// ---------------------------------------------------------------------------
__global__ __launch_bounds__(256) void norm_query_kernel(
        const float* __restrict__ query, unsigned short* __restrict__ qn) {
    const int wave = threadIdx.x >> 5, lane = threadIdx.x & 31;
    const int row  = blockIdx.x * 8 + wave;
    const f32x4* q4 = (const f32x4*)(query + (size_t)row * DIM);

    f32x4 v[4];
    float ssq = 0.f;
#pragma unroll
    for (int i = 0; i < 4; ++i) {
        v[i] = q4[lane + 32 * i];
        ssq += v[i][0]*v[i][0] + v[i][1]*v[i][1] + v[i][2]*v[i][2] + v[i][3]*v[i][3];
    }
#pragma unroll
    for (int m = 16; m >= 1; m >>= 1) ssq += __shfl_xor(ssq, m, 32);
    const float s = 1.0f / fmaxf(sqrtf(ssq), 1e-12f);

    uint2* o = (uint2*)(qn + (size_t)row * DIM);
#pragma unroll
    for (int i = 0; i < 4; ++i) {
        uint2 p;
        p.x = (unsigned)f2bf(v[i][0] * s) | ((unsigned)f2bf(v[i][1] * s) << 16);
        p.y = (unsigned)f2bf(v[i][2] * s) | ((unsigned)f2bf(v[i][3] * s) << 16);
        o[lane + 32 * i] = p;
    }
}

// ---------------------------------------------------------------------------
// Kernel 2: queue f32 -> bf16 (rows already unit norm). Grid-stride on float4.
// ---------------------------------------------------------------------------
__global__ __launch_bounds__(256) void conv_queue_kernel(
        const float* __restrict__ queue, unsigned short* __restrict__ qb, int n4) {
    int i = blockIdx.x * blockDim.x + threadIdx.x;
    const int stride = gridDim.x * blockDim.x;
    const f32x4* q4 = (const f32x4*)queue;
    uint2* o = (uint2*)qb;
    for (; i < n4; i += stride) {
        f32x4 v = q4[i];
        uint2 p;
        p.x = (unsigned)f2bf(v[0]) | ((unsigned)f2bf(v[1]) << 16);
        p.y = (unsigned)f2bf(v[2]) | ((unsigned)f2bf(v[3]) << 16);
        o[i] = p;
    }
}

// ---------------------------------------------------------------------------
// Kernel 3: fused GEMM (WMMA bf16) + streaming top-16 + softmax + gather.
// Block = 16 query rows, 8 waves; wave w owns queue tiles {w, w+8, ...}.
// ---------------------------------------------------------------------------
__global__ __launch_bounds__(256) void retrieve_topk_kernel(
        const float*          __restrict__ queue,   // f32, for exact gather
        const unsigned short* __restrict__ qn,      // normalized query bf16
        const unsigned short* __restrict__ qb,      // queue bf16
        float* __restrict__ out_retr,
        float* __restrict__ out_dist,
        float* __restrict__ out_w) {
    __shared__ float stage[8][256];        // per-wave 16x16 sim staging
    __shared__ float mvals[8][16][TOPK];   // per-wave per-row top-16 values
    __shared__ int   midx [8][16][TOPK];   // per-wave per-row top-16 indices
    __shared__ float s_w[16][TOPK];        // final softmax weights
    __shared__ int   s_i[16][TOPK];        // final indices

    const int tid  = threadIdx.x;
    const int wave = tid >> 5, lane = tid & 31;
    const int row0 = blockIdx.x * 16;
    const int mn   = lane & 15;        // A row / B column owned by this lane
    const int kgrp = lane >> 4;        // lane half selects K sub-range

    // --- Load all A fragments (16 rows x 512) into registers, contiguous layout:
    // lanes 0-15 : elem j -> K=kk+j
    // lanes 16-31: elem j -> K=kk+16+j
    bf16x16 a_frag[16];
    {
        const unsigned short* arow = qn + (size_t)(row0 + mn) * DIM;
#pragma unroll
        for (int kki = 0; kki < 16; ++kki) {
            const int kk = kki * 32 + kgrp * 16;
            ABFrag u;
            u.f[0] = *(const f32x4*)(arow + kk);
            u.f[1] = *(const f32x4*)(arow + kk + 8);
            a_frag[kki] = u.bf;
        }
    }

    float vals[TOPK]; int idxs[TOPK];
#pragma unroll
    for (int k = 0; k < TOPK; ++k) { vals[k] = NEG_INF; idxs[k] = 0; }

    // B layout: lane holds column N=mn, K = kk + 16*kgrp + (0..15) contiguous.
    const size_t lane_off = (size_t)mn * DIM + kgrp * 16;
    const unsigned short* brow0 = qb + lane_off;   // + tile*16*DIM per tile

    // Half-tile double buffers: 16 b128 loads always in flight.
    ABFrag b0[8], b1[8];

    // prologue: first half of tile (j=0)
    {
        const unsigned short* br = brow0 + (size_t)wave * 16 * DIM;
#pragma unroll
        for (int kki = 0; kki < 8; ++kki) {
            const f32x4* bp = (const f32x4*)(br + kki * 32);
            b0[kki].f[0] = bp[0];
            b0[kki].f[1] = bp[1];
        }
    }

    const int NTILES = (Q_ROWS / 16) / 8;   // 256 tiles per wave
    for (int j = 0; j < NTILES; ++j) {
        const int tile = j * 8 + wave;
        const int jn   = (j + 1 < NTILES) ? j + 1 : j;           // clamped
        const unsigned short* br_cur  = brow0 + (size_t)tile * 16 * DIM;
        const unsigned short* br_next = brow0 + (size_t)(jn * 8 + wave) * 16 * DIM;

        // Warm WGP$/L2 two tiles ahead (global_prefetch_b8).
        const int jp = (j + 2 < NTILES) ? j + 2 : j;
        __builtin_prefetch(brow0 + (size_t)(jp * 8 + wave) * 16 * DIM, 0, 3);

        // issue loads for second half of current tile (kki 8..15)
#pragma unroll
        for (int kki = 0; kki < 8; ++kki) {
            const f32x4* bp = (const f32x4*)(br_cur + (kki + 8) * 32);
            b1[kki].f[0] = bp[0];
            b1[kki].f[1] = bp[1];
        }

        v8f c = {};
#pragma unroll
        for (int kki = 0; kki < 8; ++kki)       // consume b0 (waits <=16)
            c = __builtin_amdgcn_wmma_f32_16x16x32_bf16(
                    false, a_frag[kki], false, b0[kki].bf, (short)0, c, false, false);

        // issue loads for first half of next tile (kki 0..7)
#pragma unroll
        for (int kki = 0; kki < 8; ++kki) {
            const f32x4* bp = (const f32x4*)(br_next + kki * 32);
            b0[kki].f[0] = bp[0];
            b0[kki].f[1] = bp[1];
        }

#pragma unroll
        for (int kki = 0; kki < 8; ++kki)       // consume b1 (waits <=16)
            c = __builtin_amdgcn_wmma_f32_16x16x32_bf16(
                    false, a_frag[kki + 8], false, b1[kki].bf, (short)0, c, false, false);

        // Stage sims via per-wave LDS; same-wave DS pipe is in-order, so a
        // compiler fence suffices (no block barrier -> waves run decoupled).
#pragma unroll
        for (int i = 0; i < 8; ++i)
            stage[wave][(i + (kgrp << 3)) * 16 + mn] = c[i];
        KFENCE();
        if (lane < 16) {
            const int cbase = tile * 16;
#pragma unroll
            for (int e = 0; e < 16; ++e)
                insert16(vals, idxs, stage[wave][lane * 16 + e], cbase + e);
        }
        KFENCE();
    }

    // --- Cross-wave merge: columns are disjoint across waves. ---
    if (lane < 16) {
#pragma unroll
        for (int k = 0; k < TOPK; ++k) {
            mvals[wave][lane][k] = vals[k];
            midx [wave][lane][k] = idxs[k];
        }
    }
    __syncthreads();

    if (tid < 16) {
        float fv[TOPK]; int fi[TOPK];
#pragma unroll
        for (int k = 0; k < TOPK; ++k) { fv[k] = NEG_INF; fi[k] = 0; }
        for (int w = 0; w < 8; ++w)
#pragma unroll
            for (int k = 0; k < TOPK; ++k)
                insert16(fv, fi, mvals[w][tid][k], midx[w][tid][k]);

        const float mx = fv[0];
        out_dist[row0 + tid] = 1.0f - mx;

        float e[TOPK], sum = 0.f;
#pragma unroll
        for (int k = 0; k < TOPK; ++k) {           // softmax, T = 0.1
            e[k] = __expf((fv[k] - mx) * 10.0f);
            sum += e[k];
        }
        const float inv = 1.0f / sum;
#pragma unroll
        for (int k = 0; k < TOPK; ++k) {
            const float wk = e[k] * inv;
            out_w[(size_t)(row0 + tid) * TOPK + k] = wk;
            s_w[tid][k] = wk;
            s_i[tid][k] = fi[k];
        }
    }
    __syncthreads();

    // --- Gather + weighted sum from f32 queue, then L2-normalize. ---
    const int r  = tid >> 4;        // row 0..15
    const int cc = tid & 15;        // dim phase; 16 threads cover one row
    float acc[32];
#pragma unroll
    for (int d = 0; d < 32; ++d) acc[d] = 0.f;
    for (int k = 0; k < TOPK; ++k) {
        const float wk = s_w[r][k];
        const float* qrow = queue + (size_t)s_i[r][k] * DIM;
#pragma unroll 8
        for (int d = 0; d < 32; ++d)
            acc[d] += wk * qrow[cc + d * 16];
    }
    float ssq = 0.f;
#pragma unroll
    for (int d = 0; d < 32; ++d) ssq += acc[d] * acc[d];
#pragma unroll
    for (int m = 8; m >= 1; m >>= 1) ssq += __shfl_xor(ssq, m, 32);
    const float s = 1.0f / fmaxf(sqrtf(ssq), 1e-12f);

    float* orow = out_retr + (size_t)(row0 + r) * DIM;
#pragma unroll
    for (int d = 0; d < 32; ++d) orow[cc + d * 16] = acc[d] * s;
}

// ---------------------------------------------------------------------------
extern "C" void kernel_launch(void* const* d_in, const int* in_sizes, int n_in,
                              void* d_out, int out_size, void* d_ws, size_t ws_size,
                              hipStream_t stream) {
    (void)in_sizes; (void)n_in; (void)out_size; (void)ws_size;
    const float* query = (const float*)d_in[0];   // [4096, 512]
    const float* queue = (const float*)d_in[1];   // [32768, 512]

    float* out      = (float*)d_out;
    float* out_retr = out;                                    // 4096*512
    float* out_dist = out + (size_t)B_ROWS * DIM;             // 4096
    float* out_w    = out_dist + B_ROWS;                      // 4096*16

    unsigned short* qn = (unsigned short*)d_ws;               // 4 MB
    unsigned short* qb = qn + (size_t)B_ROWS * DIM;           // 32 MB

    norm_query_kernel<<<B_ROWS / 8, 256, 0, stream>>>(query, qn);
    conv_queue_kernel<<<2048, 256, 0, stream>>>(queue, qb, Q_ROWS * DIM / 4);
    retrieve_topk_kernel<<<B_ROWS / 16, 256, 0, stream>>>(
        queue, qn, qb, out_retr, out_dist, out_w);
}
